// OFAModel_53386443489529
// MI455X (gfx1250) — matrix operand order repaired and checked by reading, not compiled, not run on hardware
//
#include <hip/hip_runtime.h>
#include <hip/hip_bf16.h>
#include <math.h>

#define N_NODES 4096
#define IN_DIM  256
#define HID     256
#define TASK    128
#define NLAYERS 2
#define NEG_INF_F (-9e15f)

typedef __attribute__((ext_vector_type(16))) __bf16 v16bf;
typedef __attribute__((ext_vector_type(8)))  float  v8f;
typedef __attribute__((ext_vector_type(4)))  int    v4i;

#define AS1 __attribute__((address_space(1)))
#define AS3 __attribute__((address_space(3)))
typedef AS1 v4i* gvec16_t;   // global 16B-vector pointer (async-LDS builtin arg0)
typedef AS3 v4i* lvec16_t;   // LDS 16B-vector pointer     (async-LDS builtin arg1)

#if defined(__has_builtin)
#  if __has_builtin(__builtin_amdgcn_global_load_async_to_lds_b128)
#    define HAVE_ASYNC_LDS 1
#  endif
#endif
#ifndef HAVE_ASYNC_LDS
#  define HAVE_ASYNC_LDS 0
#endif

__device__ __forceinline__ void wait_async0() {
#if defined(__has_builtin) && __has_builtin(__builtin_amdgcn_s_wait_asynccnt)
    __builtin_amdgcn_s_wait_asynccnt(0);
#else
    asm volatile("s_wait_asynccnt 0" ::: "memory");
#endif
}

__device__ __forceinline__ float lrelu(float x) { return x > 0.f ? x : 0.01f * x; }

__device__ __forceinline__ v8f bf16_wmma(v16bf a, v16bf b, v8f c) {
    return __builtin_amdgcn_wmma_f32_16x16x32_bf16(false, a, false, b, (short)0, c, false, false);
}

// WMMA lane swizzle (16-bit operands, wave32):
//   A (16xK): lane = row + 16*(kloc>>4),     elem = kloc & 15
//   B (Kx16): lane = col + 16*(kloc>>4),     elem = kloc & 15
// so each lane's 16-bf16 fragment is one contiguous 32-byte vector.

// ---------------------------------------------------------------------------
// Pre-swizzle a [K,256] f32 weight matrix into WMMA-B bf16 lane order.
// ---------------------------------------------------------------------------
__global__ __launch_bounds__(256)
void swizzle_weights_kernel(const float* __restrict__ src, __bf16* __restrict__ dst, int K)
{
    int e = blockIdx.x * 256 + threadIdx.x;
    if (e >= K * HID) return;
    int k = e >> 8;
    int n = e & 255;
    int kloc = k & 31;
    size_t d = ((((size_t)(k >> 5) * 16 + (n >> 4)) * 32) + (n & 15) + ((kloc >> 4) << 4)) * 16
             + (kloc & 15);
    dst[d] = (__bf16)src[e];
}

// ---------------------------------------------------------------------------
// C[4096,256] = act( [A1 | A2] @ W + bias ),  Wswz pre-swizzled bf16.
// grid = (4096/16, 256/64), block = 128 (4 waves, one 16x16 tile each)
// ---------------------------------------------------------------------------
template<int ACT>
__global__ __launch_bounds__(128)
void wmma_gemm_kernel(const float* __restrict__ A1, int K1,
                      const float* __restrict__ A2, int K2,
                      const __bf16* __restrict__ Wswz,
                      const float* __restrict__ bias,
                      float* __restrict__ C)
{
    __shared__ __align__(32) __bf16 a_swz[32][16];
    const int tid   = threadIdx.x;
    const int lane  = tid & 31;
    const int wave  = tid >> 5;
    const int hf    = lane >> 4;
    const int r0    = blockIdx.x * 16;
    const int cTile = blockIdx.y * 4 + wave;   // 16-col tile index 0..15
    const int K     = K1 + K2;

    v8f acc = {};
    for (int k0 = 0; k0 < K; k0 += 32) {
        // stage A tile directly in WMMA-A lane order
        #pragma unroll
        for (int t = 0; t < 4; ++t) {
            int e    = tid + 128 * t;          // 512 elements
            int kloc = e & 31;
            int m    = e >> 5;
            int kg   = k0 + kloc;
            float v  = (kg < K1) ? A1[(size_t)(r0 + m) * K1 + kg]
                                 : A2[(size_t)(r0 + m) * K2 + (kg - K1)];
            a_swz[m + ((kloc >> 4) << 4)][kloc & 15] = (__bf16)v;
        }
        __syncthreads();
        v16bf af  = *(const v16bf*)(&a_swz[lane][0]);
        v16bf bfr = *(const v16bf*)(Wswz + ((((size_t)(k0 >> 5) * 16 + cTile) * 32 + lane) << 4));
        acc = bf16_wmma(af, bfr, acc);
        __syncthreads();
    }

    const int n = cTile * 16 + (lane & 15);
    const float b = bias[n];
    #pragma unroll
    for (int r = 0; r < 8; ++r) {
        int row = r + 8 * hf;
        float x = acc[r] + b;
        if (ACT == 1) x = fmaxf(x, 0.f);
        if (ACT == 2) x = 1.f / (1.f + __expf(-x));
        C[(size_t)(r0 + row) * HID + n] = x;
    }
}

// ---------------------------------------------------------------------------
// gm = gate * raw (f32 + WMMA-B-swizzled bf16); accum += sum(gate)*scale
// ---------------------------------------------------------------------------
__global__ __launch_bounds__(256)
void gating_kernel(const float* __restrict__ gate, const float* __restrict__ raw,
                   float* __restrict__ gm, __bf16* __restrict__ gm_swz,
                   float* __restrict__ accum, float scale)
{
    int idx = blockIdx.x * 256 + threadIdx.x;
    float g = gate[idx];
    float v = g * raw[idx];
    gm[idx] = v;
    int i = idx >> 8, c = idx & 255;
    int kloc = i & 31;
    size_t d = ((((size_t)(i >> 5) * 16 + (c >> 4)) * 32) + (c & 15) + ((kloc >> 4) << 4)) * 16
             + (kloc & 15);
    gm_swz[d] = (__bf16)v;
    float s = g * scale;
    #pragma unroll
    for (int off = 16; off > 0; off >>= 1) s += __shfl_xor(s, off, 32);
    if ((threadIdx.x & 31) == 0) atomicAdd(accum, s);
}

// ---------------------------------------------------------------------------
// out[r] = X[r,:] . vec  (one wave per row, float4)
// ---------------------------------------------------------------------------
__global__ __launch_bounds__(256)
void rowdot_kernel(const float* __restrict__ X, const float* __restrict__ vec,
                   float* __restrict__ out)
{
    int lane = threadIdx.x & 31;
    int wave = threadIdx.x >> 5;
    int row  = blockIdx.x * 8 + wave;
    const float4* xr = (const float4*)(X + (size_t)row * HID);
    const float4* vr = (const float4*)vec;
    float s = 0.f;
    #pragma unroll
    for (int c4 = lane; c4 < HID / 4; c4 += 32) {
        float4 x = xr[c4];
        float4 v = vr[c4];
        s += x.x * v.x + x.y * v.y + x.z * v.z + x.w * v.w;
    }
    #pragma unroll
    for (int off = 16; off > 0; off >>= 1) s += __shfl_xor(s, off, 32);
    if (lane == 0) out[row] = s;
}

// ---------------------------------------------------------------------------
// Per-row masked online softmax stats (one wave per row, int4 adj + prefetch)
// ---------------------------------------------------------------------------
#define ONLINE_UPD(vv)                                            \
    do { float _v = (vv);                                         \
         if (_v > m) { s = s * __expf(m - _v) + 1.f; m = _v; }    \
         else        { s += __expf(_v - m); } } while (0)

__global__ __launch_bounds__(256)
void softmax_stats_kernel(const int* __restrict__ adj, const float* __restrict__ qq,
                          const float* __restrict__ kk,
                          float* __restrict__ mrow, float* __restrict__ sinv)
{
    int lane = threadIdx.x & 31;
    int wave = threadIdx.x >> 5;
    int i    = blockIdx.x * 8 + wave;
    float qi = qq[i];
    float m = -INFINITY, s = 0.f;
    const int4*   arow = (const int4*)(adj + (size_t)i * N_NODES);
    const float4* krow = (const float4*)kk;
    for (int jb = lane; jb < N_NODES / 4; jb += 32) {
        int4   av = arow[jb];
        float4 kv = krow[jb];
        __builtin_prefetch((const void*)(arow + jb + 64), 0, 0);
        ONLINE_UPD(av.x > 0 ? lrelu(qi + kv.x) : NEG_INF_F);
        ONLINE_UPD(av.y > 0 ? lrelu(qi + kv.y) : NEG_INF_F);
        ONLINE_UPD(av.z > 0 ? lrelu(qi + kv.z) : NEG_INF_F);
        ONLINE_UPD(av.w > 0 ? lrelu(qi + kv.w) : NEG_INF_F);
    }
    #pragma unroll
    for (int off = 16; off > 0; off >>= 1) {
        float mo = __shfl_xor(m, off, 32);
        float so = __shfl_xor(s, off, 32);
        float mn = fmaxf(m, mo);
        s = s * __expf(m - mn) + so * __expf(mo - mn);
        m = mn;
    }
    if (lane == 0) { mrow[i] = m; sinv[i] = 1.f / s; }
}

// ---------------------------------------------------------------------------
// hout[j,:] = 0.5*( sum_i att[i,j]*gm[i,:] + hold[j,:] )
// grid = 4096/16 (16 j-rows per block), block = 256 (8 waves x 32-col slab).
// Attention tile built branchless in WMMA-A lane order; gm read pre-swizzled.
// Stats vectors staged to LDS with async-load-to-LDS when available.
// ---------------------------------------------------------------------------
__global__ __launch_bounds__(256)
void agg_kernel(const int* __restrict__ adj, const __bf16* __restrict__ gm_swz,
                const float* __restrict__ qq, const float* __restrict__ kk,
                const float* __restrict__ mrow, const float* __restrict__ sinv,
                const float* __restrict__ hold, float* __restrict__ hout)
{
    __shared__ __align__(32) __bf16 w_swz[32][16];
    __shared__ __align__(16) float qv[32];
    __shared__ __align__(16) float mv[32];
    __shared__ __align__(16) float sv[32];
    __shared__ __align__(16) float kk16[16];
    const int tid  = threadIdx.x;
    const int lane = tid & 31;
    const int wave = tid >> 5;
    const int hf   = lane >> 4;
    const int j0   = blockIdx.x * 16;

    // stage this block's kk values once
#if HAVE_ASYNC_LDS
    if (tid < 4)
        __builtin_amdgcn_global_load_async_to_lds_b128(
            (gvec16_t)(kk + j0 + tid * 4), (lvec16_t)(kk16 + tid * 4), 0, 0);
#else
    if (tid < 16) kk16[tid] = kk[j0 + tid];
#endif

    v8f acc0 = {}, acc1 = {};
    for (int i0 = 0; i0 < N_NODES; i0 += 32) {
        // ---- stage per-chunk stats (qq/mrow/sinv for 32 i's)
#if HAVE_ASYNC_LDS
        if (tid < 24) {
            int g = tid >> 3, t8 = tid & 7;
            const float* gsrc = (g == 0) ? qq : ((g == 1) ? mrow : sinv);
            float*       ldst = (g == 0) ? qv : ((g == 1) ? mv : sv);
            __builtin_amdgcn_global_load_async_to_lds_b128(
                (gvec16_t)(gsrc + i0 + t8 * 4), (lvec16_t)(ldst + t8 * 4), 0, 0);
        }
        wait_async0();
#else
        if (tid < 32)       qv[tid]      = qq[i0 + tid];
        else if (tid < 64)  mv[tid - 32] = mrow[i0 + tid - 32];
        else if (tid < 96)  sv[tid - 64] = sinv[i0 + tid - 64];
#endif
        __syncthreads();

        // ---- attention^T tile (16 j x 32 i) in WMMA-A lane order, branchless
        {
            int il = tid >> 3;               // i_loc 0..31
            int j2 = (tid & 7) << 1;         // j_loc pair base
            int2 av = *(const int2*)(adj + (size_t)(i0 + il) * N_NODES + j0 + j2);
            if (i0 + 32 < N_NODES)
                __builtin_prefetch((const void*)(adj + (size_t)(i0 + 32 + il) * N_NODES + j0 + j2), 0, 0);
            float qi = qv[il], mi = mv[il], si = sv[il];
            float e0 = lrelu(qi + kk16[j2]);
            float e1 = lrelu(qi + kk16[j2 + 1]);
            float v0 = (av.x > 0) ? e0 : NEG_INF_F;
            float v1 = (av.y > 0) ? e1 : NEG_INF_F;
            float w0 = __expf(v0 - mi) * si;
            float w1 = __expf(v1 - mi) * si;
            int lnh = (il >> 4) << 4;
            int tt  = il & 15;
            w_swz[j2 + lnh][tt]     = (__bf16)w0;
            w_swz[j2 + 1 + lnh][tt] = (__bf16)w1;
        }
        __syncthreads();

        // ---- fragments: contiguous 32B vector loads
        v16bf af = *(const v16bf*)(&w_swz[lane][0]);
        size_t cb = (size_t)(i0 >> 5) * 16 * 32 * 16;
        v16bf b0 = *(const v16bf*)(gm_swz + cb + (((size_t)(wave * 2)     * 32 + lane) << 4));
        v16bf b1 = *(const v16bf*)(gm_swz + cb + (((size_t)(wave * 2 + 1) * 32 + lane) << 4));
        acc0 = bf16_wmma(af, b0, acc0);
        acc1 = bf16_wmma(af, b1, acc1);
        __syncthreads();
    }

    // ---- epilogue: h = (agg + h)/2
    const int nc = lane & 15;
    #pragma unroll
    for (int r = 0; r < 8; ++r) {
        int j = j0 + r + 8 * hf;
        size_t idx0 = (size_t)j * HID + wave * 32 + nc;
        size_t idx1 = idx0 + 16;
        hout[idx0] = 0.5f * (acc0[r] + hold[idx0]);
        hout[idx1] = 0.5f * (acc1[r] + hold[idx1]);
    }
}

__global__ void finalize_kernel(const float* __restrict__ accum, float* __restrict__ dst)
{
    if (threadIdx.x == 0) dst[0] = accum[0];
}

// ---------------------------------------------------------------------------
extern "C" void kernel_launch(void* const* d_in, const int* in_sizes, int n_in,
                              void* d_out, int out_size, void* d_ws, size_t ws_size,
                              hipStream_t stream)
{
    (void)in_sizes; (void)n_in; (void)out_size; (void)ws_size;

    const float* h   = (const float*)d_in[0];
    const int*   adj = (const int*)  d_in[1];
    const float* z   = (const float*)d_in[2];
    const float* Wp  = (const float*)d_in[3];
    const float* bp  = (const float*)d_in[4];
    const float* Wm  = (const float*)d_in[5];
    const float* bm  = (const float*)d_in[6];
    const float* Wg  = (const float*)d_in[7];
    const float* bg  = (const float*)d_in[8];
    const float* Wk  = (const float*)d_in[9];
    const float* bk  = (const float*)d_in[10];
    const float* Wq  = (const float*)d_in[11];
    const float* bq  = (const float*)d_in[12];
    const float* a   = (const float*)d_in[13];

    float* out = (float*)d_out;
    float* ws  = (float*)d_ws;
    const size_t M = (size_t)N_NODES * HID;

    float* hcur  = ws;
    float* gm    = ws + M;
    float* bufA  = ws + 2 * M;   // raw message, later k
    float* bufB  = ws + 3 * M;   // gate, later q
    float* qq    = ws + 4 * M;
    float* kk    = ws + 4 * M + N_NODES;
    float* mrow  = ws + 4 * M + 2 * N_NODES;
    float* sinv  = ws + 4 * M + 3 * N_NODES;
    float* accum = ws + 4 * M + 4 * N_NODES;
    __bf16* gm_swz = (__bf16*)(ws + 4 * M + 4 * (size_t)N_NODES + 16);  // M bf16
    __bf16* wbf    = gm_swz + M;                                        // swizzled weights

    const size_t WP_SZ = (size_t)IN_DIM * HID;          // 65536
    const size_t WM_SZ = (size_t)HID * HID;             // 65536
    const size_t WG_SZ = (size_t)(HID + TASK) * HID;    // 98304
    __bf16* wp_bf = wbf;
    __bf16* wm_bf = wp_bf + WP_SZ;
    __bf16* wg_bf = wm_bf + NLAYERS * WM_SZ;
    __bf16* wk_bf = wg_bf + NLAYERS * WG_SZ;
    __bf16* wq_bf = wk_bf + NLAYERS * WM_SZ;

    (void)hipMemsetAsync(accum, 0, sizeof(float), stream);

    // pre-swizzle all weights into WMMA-B bf16 lane order
    swizzle_weights_kernel<<<(unsigned)(WP_SZ / 256), 256, 0, stream>>>(Wp, wp_bf, IN_DIM);
    for (int l = 0; l < NLAYERS; ++l) {
        swizzle_weights_kernel<<<(unsigned)(WM_SZ / 256), 256, 0, stream>>>(
            Wm + l * WM_SZ, wm_bf + l * WM_SZ, HID);
        swizzle_weights_kernel<<<(unsigned)(WG_SZ / 256), 256, 0, stream>>>(
            Wg + l * WG_SZ, wg_bf + l * WG_SZ, HID + TASK);
        swizzle_weights_kernel<<<(unsigned)(WM_SZ / 256), 256, 0, stream>>>(
            Wk + l * WM_SZ, wk_bf + l * WM_SZ, HID);
        swizzle_weights_kernel<<<(unsigned)(WM_SZ / 256), 256, 0, stream>>>(
            Wq + l * WM_SZ, wq_bf + l * WM_SZ, HID);
    }

    dim3 gGemm(N_NODES / 16, HID / 64);
    dim3 bGemm(128);

    // initial projection: hcur = h @ Wp + bp
    wmma_gemm_kernel<0><<<gGemm, bGemm, 0, stream>>>(h, IN_DIM, (const float*)nullptr, 0,
                                                     wp_bf, bp, hcur);

    const float scale = 1.0f / ((float)M * (float)NLAYERS);
    for (int l = 0; l < NLAYERS; ++l) {
        // raw = relu(h @ Wm + bm)
        wmma_gemm_kernel<1><<<gGemm, bGemm, 0, stream>>>(hcur, HID, (const float*)nullptr, 0,
                                                         wm_bf + l * WM_SZ, bm + l * HID, bufA);
        // gate = sigmoid([h|z] @ Wg + bg)
        wmma_gemm_kernel<2><<<gGemm, bGemm, 0, stream>>>(hcur, HID, z, TASK,
                                                         wg_bf + l * WG_SZ, bg + l * HID, bufB);
        // gm = gate * raw (+ swizzled bf16 copy, gate-L1 accumulation)
        gating_kernel<<<(unsigned)(M / 256), 256, 0, stream>>>(bufB, bufA, gm, gm_swz,
                                                               accum, scale);
        // q = gm @ Wq + bq ; k = h @ Wk + bk
        wmma_gemm_kernel<0><<<gGemm, bGemm, 0, stream>>>(gm, HID, (const float*)nullptr, 0,
                                                         wq_bf + l * WM_SZ, bq + l * HID, bufB);
        wmma_gemm_kernel<0><<<gGemm, bGemm, 0, stream>>>(hcur, HID, (const float*)nullptr, 0,
                                                         wk_bf + l * WM_SZ, bk + l * HID, bufA);
        // qq = q . a[d:], kk = k . a[:d]
        const float* a_l = a + (size_t)l * 2 * HID;
        rowdot_kernel<<<N_NODES / 8, 256, 0, stream>>>(bufB, a_l + HID, qq);
        rowdot_kernel<<<N_NODES / 8, 256, 0, stream>>>(bufA, a_l,       kk);
        // per-row masked online softmax stats
        softmax_stats_kernel<<<N_NODES / 8, 256, 0, stream>>>(adj, qq, kk, mrow, sinv);
        // aggregation + residual halving
        float* dst = (l == NLAYERS - 1) ? out : hcur;
        agg_kernel<<<N_NODES / 16, 256, 0, stream>>>(adj, gm_swz, qq, kk, mrow, sinv, hcur, dst);
    }

    finalize_kernel<<<1, 32, 0, stream>>>(accum, out + M);
}
